// NestedCell3_59493886984655
// MI455X (gfx1250) — matrix operand order repaired and checked by reading, not compiled, not run on hardware
//
#include <hip/hip_runtime.h>

// ---------------------------------------------------------------------------
// Problem constants (match reference)
// ---------------------------------------------------------------------------
#define BB   2
#define NN   2048
#define FF   64
#define HH   4
#define CC   32
#define HID  128
#define HIN  128   // H*C

typedef __attribute__((ext_vector_type(16))) __bf16        v16bf;
typedef __attribute__((ext_vector_type(8)))  float         v8f;
typedef __attribute__((ext_vector_type(4)))  unsigned int  v4u;
typedef __attribute__((ext_vector_type(4)))  float         v4f;

union FragAB {              // one WMMA 16x32 (A) or 32x16 (B) bf16 operand
    v16bf          v;
    __bf16         b[16];
    v4u            q[2];
    unsigned int   u[8];
    unsigned short s[16];
};

// Native converts: let clang use the BF16 conversion hardware (packed cvt),
// instead of a manual bfe/add3/mov_b16 RNE chain.
__device__ __forceinline__ __bf16 f2bf(float f) { return (__bf16)f; }
__device__ __forceinline__ void store_bf16(unsigned short* p, float f) {
    __bf16 b = (__bf16)f;
    *p = __builtin_bit_cast(unsigned short, b);
}
__device__ __forceinline__ float bf2f(unsigned short s) {
    unsigned int u = ((unsigned int)s) << 16;
    return __builtin_bit_cast(float, u);
}
__device__ __forceinline__ float sigmoidf_(float x) {
    return 1.0f / (1.0f + __expf(-x));
}
__device__ __forceinline__ v8f wmma_bf16(const FragAB& A, const FragAB& Bm, v8f C) {
    return __builtin_amdgcn_wmma_f32_16x16x32_bf16(
        /*neg_a=*/false, A.v, /*neg_b=*/false, Bm.v,
        /*c_mod=*/(short)0, C, /*reuse_a=*/false, /*reuse_b=*/false);
}
__device__ __forceinline__ v8f vzero8() {
    v8f z = {0.f,0.f,0.f,0.f,0.f,0.f,0.f,0.f};
    return z;
}

// ---------------------------------------------------------------------------
// K0: precompute bf16 transposed weights:
//   kernelT[hc][f]  <- kernel[f][hc]            (128 x 64)
//   Wt[w][o][i]     <- W_w[i][o]                (3 x 128 x 256)
//   Rt[o][i]        <- R_p[i][o]                (128 x 128)
// ---------------------------------------------------------------------------
__global__ __launch_bounds__(256)
void k_prep(const float* __restrict__ kern, const float* __restrict__ Wu,
            const float* __restrict__ Wr,   const float* __restrict__ Wc,
            const float* __restrict__ Rp,
            unsigned short* __restrict__ kernelT,
            unsigned short* __restrict__ Wt,
            unsigned short* __restrict__ Rt)
{
    int t = blockIdx.x * blockDim.x + threadIdx.x;
    if (t < 8192) {
        int hc = t >> 6, f = t & 63;
        store_bf16(&kernelT[t], kern[f * HIN + hc]);
    } else if (t < 8192 + 98304) {
        int u = t - 8192;
        int w = u / 32768, r = u % 32768;
        int o = r >> 8, i = r & 255;
        const float* W = (w == 0) ? Wu : ((w == 1) ? Wr : Wc);
        store_bf16(&Wt[w * 32768 + o * 256 + i], W[i * HID + o]);
    } else if (t < 122880) {
        int u = t - 106496;
        int o = u >> 7, i = u & 127;
        store_bf16(&Rt[o * HID + i], Rp[i * HID + o]);
    }
}

// ---------------------------------------------------------------------------
// K1: xp = x @ kernel   -> stored transposed as xpB[b][hc][n] (bf16)
// block = 128 threads (4 waves); block owns 16 rows x 128 cols; wave w owns
// column tiles {32w, 32w+16}; K = 64 (2 chunks).
// ---------------------------------------------------------------------------
__global__ __launch_bounds__(128)
void k_xp(const float* __restrict__ x, const unsigned short* __restrict__ kernelT,
          unsigned short* __restrict__ xpB)
{
    const int lane = threadIdx.x & 31, wave = threadIdx.x >> 5;
    const int row = lane & 15, half = lane >> 4;
    const int koffA = half * 16, koffB = half * 16;
    const int grow = blockIdx.x * 16 + row;            // 0..4095
    const int b  = (blockIdx.x * 16) >> 11;
    const int n0 = (blockIdx.x * 16) & (NN - 1);
    const int c0 = wave * 32 + (lane & 15), c1 = c0 + 16;

    v8f acc0 = vzero8(), acc1 = vzero8();
    const unsigned int* kb = (const unsigned int*)kernelT;

    #pragma unroll
    for (int kc = 0; kc < FF; kc += 32) {
        FragAB a;
        const float* xr = x + (size_t)grow * FF + kc + koffA;
        v4f x0 = *(const v4f*)(xr),      x1 = *(const v4f*)(xr + 4);
        v4f x2 = *(const v4f*)(xr + 8), x3 = *(const v4f*)(xr + 12);
        #pragma unroll
        for (int t = 0; t < 4; ++t) {
            a.b[t]      = f2bf(x0[t]);
            a.b[4 + t]  = f2bf(x1[t]);
            a.b[8 + t]  = f2bf(x2[t]);
            a.b[12 + t] = f2bf(x3[t]);
        }
        FragAB b0, b1;
        const v4u* p0 = (const v4u*)(kb + ((c0 * FF + kc + koffB) >> 1));
        const v4u* p1 = (const v4u*)(kb + ((c1 * FF + kc + koffB) >> 1));
        b0.q[0] = p0[0]; b0.q[1] = p0[1];
        b1.q[0] = p1[0]; b1.q[1] = p1[1];
        acc0 = wmma_bf16(a, b0, acc0);
        acc1 = wmma_bf16(a, b1, acc1);
    }
    #pragma unroll
    for (int v = 0; v < 8; ++v) {
        int M = v + half * 8;
        store_bf16(&xpB[((size_t)(b * HIN) + c0) * NN + n0 + M], acc0[v]);
        store_bf16(&xpB[((size_t)(b * HIN) + c1) * NN + n0 + M], acc1[v]);
    }
}

// ---------------------------------------------------------------------------
// K1b: es[b][h][n], en[b][h][n] from xpB (feature-major -> coalesced over n)
// ---------------------------------------------------------------------------
__global__ __launch_bounds__(256)
void k_scores(const unsigned short* __restrict__ xpB,
              const float* __restrict__ attn_self, const float* __restrict__ attn_nei,
              float* __restrict__ es, float* __restrict__ en)
{
    int t = blockIdx.x * blockDim.x + threadIdx.x;     // B*H*N
    if (t >= BB * HH * NN) return;
    int n = t & (NN - 1), h = (t >> 11) & 3, b = t >> 13;
    float s = 0.f, e = 0.f;
    #pragma unroll
    for (int c = 0; c < CC; ++c) {
        float v = bf2f(xpB[((size_t)(b * HIN) + h * CC + c) * NN + n]);
        s += v * attn_self[h * CC + c];
        e += v * attn_nei [h * CC + c];
    }
    es[((size_t)b * HH + h) * NN + n] = s;
    en[((size_t)b * HH + h) * NN + n] = e;
}

// ---------------------------------------------------------------------------
// K1c: z[b][n][128..255] = bf16(h[b][n][0..127])  (first half filled by attn)
// ---------------------------------------------------------------------------
__global__ __launch_bounds__(256)
void k_hz(const float* __restrict__ h, unsigned short* __restrict__ z)
{
    int t = blockIdx.x * blockDim.x + threadIdx.x;     // B*N*HID
    if (t >= BB * NN * HID) return;
    int k = t & 127, bn = t >> 7;
    store_bf16(&z[(size_t)bn * 256 + HID + k], h[t]);
}

// ---------------------------------------------------------------------------
// K2: fused GAT attention (two-pass streaming softmax + WMMA aggregation).
// block = 128 thr (4 waves); block owns (b, 16-row i-tile); wave w = head w.
// Per 32-j chunk each lane computes exactly its 16 A-layout probabilities.
// ---------------------------------------------------------------------------
__device__ __forceinline__ void attn_e16(const float* __restrict__ enh,
                                         const float* __restrict__ arow,
                                         float es_row, int j0, int koffA,
                                         float ev[16])
{
    const float* p = enh + j0 + koffA;
    const float* q = arow + j0 + koffA;
    v4f e0 = *(const v4f*)(p),      e1 = *(const v4f*)(p + 4);
    v4f e2 = *(const v4f*)(p + 8), e3 = *(const v4f*)(p + 12);
    v4f a0 = *(const v4f*)(q),      a1 = *(const v4f*)(q + 4);
    v4f a2 = *(const v4f*)(q + 8), a3 = *(const v4f*)(q + 12);
    float env[16] = {e0[0],e0[1],e0[2],e0[3], e1[0],e1[1],e1[2],e1[3],
                     e2[0],e2[1],e2[2],e2[3], e3[0],e3[1],e3[2],e3[3]};
    float av [16] = {a0[0],a0[1],a0[2],a0[3], a1[0],a1[1],a1[2],a1[3],
                     a2[0],a2[1],a2[2],a2[3], a3[0],a3[1],a3[2],a3[3]};
    #pragma unroll
    for (int t = 0; t < 16; ++t) {
        float e = es_row + env[t];
        e = (e > 0.f) ? e : 0.2f * e;            // leaky_relu(0.2)
        e = fmaf(-1.0e9f, 1.0f - av[t], e);      // adjacency mask
        ev[t] = e;
    }
}

__global__ __launch_bounds__(128)
void k_attn(const float* __restrict__ a, const float* __restrict__ es,
            const float* __restrict__ en, const float* __restrict__ bias_gat,
            const unsigned short* __restrict__ xpB, unsigned short* __restrict__ z)
{
    __shared__ float rmax[HH][16];
    __shared__ float rsum[HH][16];
    const int lane = threadIdx.x & 31, h = threadIdx.x >> 5;
    const int row = lane & 15, half = lane >> 4;
    const int koffA = half * 16, koffB = half * 16;
    const int b = blockIdx.y, i0 = blockIdx.x * 16;

    const float  es_row = es[((size_t)b * HH + h) * NN + i0 + row];
    const float* enh    = en + ((size_t)b * HH + h) * NN;
    const float* arow   = a + ((size_t)(b * NN) + i0 + row) * NN;

    // ---- pass 1: row max over masked scores -------------------------------
    float m = -3.0e38f;
    for (int j0 = 0; j0 < NN; j0 += 32) {
        float ev[16];
        attn_e16(enh, arow, es_row, j0, koffA, ev);
        #pragma unroll
        for (int t = 0; t < 16; ++t) m = fmaxf(m, ev[t]);
    }
    m = fmaxf(m, __shfl_xor(m, 16));
    if (half == 0) rmax[h][row] = m;
    __syncthreads();
    const float rm = rmax[h][row];

    // ---- pass 2: unnormalized exp -> WMMA accumulate + row sums -----------
    v8f acc0 = vzero8(), acc1 = vzero8();
    float lsum = 0.f;
    const int col = h * CC + (lane & 15);
    const unsigned int* xb = (const unsigned int*)xpB;
    const size_t bidx0 = ((size_t)(b * HIN) + col) * NN;
    const size_t bidx1 = ((size_t)(b * HIN) + col + 16) * NN;

    for (int j0 = 0; j0 < NN; j0 += 32) {
        __builtin_prefetch(arow + j0 + 256, 0, 0);
        float ev[16];
        attn_e16(enh, arow, es_row, j0, koffA, ev);
        FragAB pa;
        #pragma unroll
        for (int t = 0; t < 16; ++t) {
            float p = __expf(ev[t] - rm);
            lsum += p;
            pa.b[t] = f2bf(p);
        }
        FragAB b0, b1;
        const v4u* p0 = (const v4u*)(xb + ((bidx0 + j0 + koffB) >> 1));
        const v4u* p1 = (const v4u*)(xb + ((bidx1 + j0 + koffB) >> 1));
        b0.q[0] = p0[0]; b0.q[1] = p0[1];
        b1.q[0] = p1[0]; b1.q[1] = p1[1];
        acc0 = wmma_bf16(pa, b0, acc0);
        acc1 = wmma_bf16(pa, b1, acc1);
    }
    lsum += __shfl_xor(lsum, 16);
    if (half == 0) rsum[h][row] = lsum;
    __syncthreads();

    // ---- epilogue: normalize + bias, write conv_u (bf16) into z[:,:,0:128]
    #pragma unroll
    for (int v = 0; v < 8; ++v) {
        int M = v + half * 8;
        float inv = 1.0f / rsum[h][M];
        int c0 = h * CC + (lane & 15), c1 = c0 + 16;
        float o0 = acc0[v] * inv + bias_gat[c0];
        float o1 = acc1[v] * inv + bias_gat[c1];
        size_t rbase = ((size_t)(b * NN) + i0 + M) * 256;
        store_bf16(&z[rbase + c0], o0);
        store_bf16(&z[rbase + c1], o1);
    }
}

// ---------------------------------------------------------------------------
// K3: fused GRU gate. block = 128 thr (4 waves), block owns (b, 16-row tile).
// u,r: z @ W (K=256). c: first half from z, second half from r*h staged in LDS.
// ---------------------------------------------------------------------------
__global__ __launch_bounds__(128)
void k_gru(const unsigned short* __restrict__ z, const unsigned short* __restrict__ Wt,
           const float* __restrict__ b_u, const float* __restrict__ b_r,
           const float* __restrict__ b_c, const float* __restrict__ h,
           float* __restrict__ out_hp, unsigned short* __restrict__ hpbf)
{
    __shared__ __align__(16) unsigned short rh[16][HID];
    const int lane = threadIdx.x & 31, wave = threadIdx.x >> 5;
    const int row = lane & 15, half = lane >> 4;
    const int koffA = half * 16, koffB = half * 16;
    const int b = blockIdx.y, i0 = blockIdx.x * 16;
    const int c0 = wave * 32 + (lane & 15), c1 = c0 + 16;

    const unsigned int* wu = (const unsigned int*)(Wt);
    const unsigned int* wr = (const unsigned int*)(Wt + 32768);
    const unsigned int* wc = (const unsigned int*)(Wt + 65536);
    const unsigned short* zr = z + ((size_t)(b * NN) + i0 + row) * 256;

    v8f aU0 = vzero8(), aU1 = vzero8(), aR0 = vzero8(), aR1 = vzero8();
    v8f aC0 = vzero8(), aC1 = vzero8();

    #pragma unroll
    for (int kc = 0; kc < 256; kc += 32) {
        FragAB aF;
        aF.q[0] = *(const v4u*)(zr + kc + koffA);
        aF.q[1] = *(const v4u*)(zr + kc + koffA + 8);
        FragAB u0, u1, r0, r1;
        u0.q[0] = ((const v4u*)(wu + ((c0 * 256 + kc + koffB) >> 1)))[0];
        u0.q[1] = ((const v4u*)(wu + ((c0 * 256 + kc + koffB) >> 1)))[1];
        u1.q[0] = ((const v4u*)(wu + ((c1 * 256 + kc + koffB) >> 1)))[0];
        u1.q[1] = ((const v4u*)(wu + ((c1 * 256 + kc + koffB) >> 1)))[1];
        r0.q[0] = ((const v4u*)(wr + ((c0 * 256 + kc + koffB) >> 1)))[0];
        r0.q[1] = ((const v4u*)(wr + ((c0 * 256 + kc + koffB) >> 1)))[1];
        r1.q[0] = ((const v4u*)(wr + ((c1 * 256 + kc + koffB) >> 1)))[0];
        r1.q[1] = ((const v4u*)(wr + ((c1 * 256 + kc + koffB) >> 1)))[1];
        aU0 = wmma_bf16(aF, u0, aU0);  aU1 = wmma_bf16(aF, u1, aU1);
        aR0 = wmma_bf16(aF, r0, aR0);  aR1 = wmma_bf16(aF, r1, aR1);
        if (kc < 128) {                            // uniform branch
            FragAB cB0, cB1;
            cB0.q[0] = ((const v4u*)(wc + ((c0 * 256 + kc + koffB) >> 1)))[0];
            cB0.q[1] = ((const v4u*)(wc + ((c0 * 256 + kc + koffB) >> 1)))[1];
            cB1.q[0] = ((const v4u*)(wc + ((c1 * 256 + kc + koffB) >> 1)))[0];
            cB1.q[1] = ((const v4u*)(wc + ((c1 * 256 + kc + koffB) >> 1)))[1];
            aC0 = wmma_bf16(aF, cB0, aC0);  aC1 = wmma_bf16(aF, cB1, aC1);
        }
    }

    // r = sigmoid(b_r + .) ; stage r*h (bf16) in LDS for the c-GEMM tail
    #pragma unroll
    for (int v = 0; v < 8; ++v) {
        int M = v + half * 8, gi = i0 + M;
        float br = b_r[gi];
        float r0 = sigmoidf_(br + aR0[v]);
        float r1 = sigmoidf_(br + aR1[v]);
        size_t hb = ((size_t)(b * NN) + gi) * HID;
        store_bf16(&rh[M][c0], r0 * h[hb + c0]);
        store_bf16(&rh[M][c1], r1 * h[hb + c1]);
    }
    __syncthreads();

    #pragma unroll
    for (int kc = 0; kc < 128; kc += 32) {
        FragAB aF;
        aF.q[0] = *(const v4u*)(&rh[row][kc + koffA]);
        aF.q[1] = *(const v4u*)(&rh[row][kc + koffA + 8]);
        FragAB cB0, cB1;
        cB0.q[0] = ((const v4u*)(wc + ((c0 * 256 + 128 + kc + koffB) >> 1)))[0];
        cB0.q[1] = ((const v4u*)(wc + ((c0 * 256 + 128 + kc + koffB) >> 1)))[1];
        cB1.q[0] = ((const v4u*)(wc + ((c1 * 256 + 128 + kc + koffB) >> 1)))[0];
        cB1.q[1] = ((const v4u*)(wc + ((c1 * 256 + 128 + kc + koffB) >> 1)))[1];
        aC0 = wmma_bf16(aF, cB0, aC0);  aC1 = wmma_bf16(aF, cB1, aC1);
    }

    #pragma unroll
    for (int v = 0; v < 8; ++v) {
        int M = v + half * 8, gi = i0 + M;
        float bu = b_u[gi], bc = b_c[gi];
        float u0 = sigmoidf_(bu + aU0[v]);
        float u1 = sigmoidf_(bu + aU1[v]);
        float cc0 = tanhf(bc + aC0[v]);
        float cc1 = tanhf(bc + aC1[v]);
        size_t hb = ((size_t)(b * NN) + gi) * HID;
        float h0 = h[hb + c0], h1 = h[hb + c1];
        float p0 = u0 * h0 + (1.f - u0) * cc0;
        float p1 = u1 * h1 + (1.f - u1) * cc1;
        out_hp[hb + c0] = p0;             out_hp[hb + c1] = p1;
        store_bf16(&hpbf[hb + c0], p0);   store_bf16(&hpbf[hb + c1], p1);
    }
}

// ---------------------------------------------------------------------------
// K4: g = h' @ R_p  (4096x128 @ 128x128), g stored row-major bf16.
// ---------------------------------------------------------------------------
__global__ __launch_bounds__(128)
void k_gp(const unsigned short* __restrict__ hpbf, const unsigned short* __restrict__ Rt,
          unsigned short* __restrict__ g)
{
    const int lane = threadIdx.x & 31, wave = threadIdx.x >> 5;
    const int row = lane & 15, half = lane >> 4;
    const int koffA = half * 16, koffB = half * 16;
    const int b = blockIdx.y, i0 = blockIdx.x * 16;
    const int c0 = wave * 32 + (lane & 15), c1 = c0 + 16;

    const unsigned short* ar = hpbf + ((size_t)(b * NN) + i0 + row) * HID;
    const unsigned int* rt = (const unsigned int*)Rt;
    v8f acc0 = vzero8(), acc1 = vzero8();

    #pragma unroll
    for (int kc = 0; kc < HID; kc += 32) {
        FragAB aF;
        aF.q[0] = *(const v4u*)(ar + kc + koffA);
        aF.q[1] = *(const v4u*)(ar + kc + koffA + 8);
        FragAB b0, b1;
        b0.q[0] = ((const v4u*)(rt + ((c0 * HID + kc + koffB) >> 1)))[0];
        b0.q[1] = ((const v4u*)(rt + ((c0 * HID + kc + koffB) >> 1)))[1];
        b1.q[0] = ((const v4u*)(rt + ((c1 * HID + kc + koffB) >> 1)))[0];
        b1.q[1] = ((const v4u*)(rt + ((c1 * HID + kc + koffB) >> 1)))[1];
        acc0 = wmma_bf16(aF, b0, acc0);
        acc1 = wmma_bf16(aF, b1, acc1);
    }
    #pragma unroll
    for (int v = 0; v < 8; ++v) {
        int M = v + half * 8;
        size_t rb = ((size_t)(b * NN) + i0 + M) * HID;
        store_bf16(&g[rb + c0], acc0[v]);
        store_bf16(&g[rb + c1], acc1[v]);
    }
}

// ---------------------------------------------------------------------------
// K5: bilinear decode A[b,i,j] = sum_k g[b,i,k] * h'[b,j,k].
// block = 128 thr; wave w owns a 16x16 output tile; K = 128 (4 chunks).
// h' row-major [j][k] is directly the B operand.
// ---------------------------------------------------------------------------
__global__ __launch_bounds__(128)
void k_dec(const unsigned short* __restrict__ g, const unsigned short* __restrict__ hpbf,
           float* __restrict__ outA)
{
    const int lane = threadIdx.x & 31, wave = threadIdx.x >> 5;
    const int row = lane & 15, half = lane >> 4;
    const int koffA = half * 16, koffB = half * 16;
    const int b = blockIdx.z, i0 = blockIdx.x * 16;
    const int j0 = blockIdx.y * 64 + wave * 16;
    const int jcol = j0 + (lane & 15);

    const unsigned short* ar = g + ((size_t)(b * NN) + i0 + row) * HID;
    const unsigned int*   hb = (const unsigned int*)hpbf;
    const size_t bbase = ((size_t)(b * NN) + jcol) * HID;

    v8f acc = vzero8();
    #pragma unroll
    for (int kc = 0; kc < HID; kc += 32) {
        FragAB aF;
        aF.q[0] = *(const v4u*)(ar + kc + koffA);
        aF.q[1] = *(const v4u*)(ar + kc + koffA + 8);
        FragAB bF;
        const v4u* bp = (const v4u*)(hb + ((bbase + kc + koffB) >> 1));
        bF.q[0] = bp[0]; bF.q[1] = bp[1];
        acc = wmma_bf16(aF, bF, acc);
    }
    #pragma unroll
    for (int v = 0; v < 8; ++v) {
        int M = v + half * 8;
        outA[((size_t)(b * NN) + i0 + M) * NN + jcol] = acc[v];
    }
}

// ---------------------------------------------------------------------------
// Launch
// ---------------------------------------------------------------------------
extern "C" void kernel_launch(void* const* d_in, const int* in_sizes, int n_in,
                              void* d_out, int out_size, void* d_ws, size_t ws_size,
                              hipStream_t stream)
{
    const float* x         = (const float*)d_in[0];
    const float* a         = (const float*)d_in[1];
    const float* h         = (const float*)d_in[2];
    const float* kern      = (const float*)d_in[3];
    const float* attn_self = (const float*)d_in[4];
    const float* attn_nei  = (const float*)d_in[5];
    const float* bias_gat  = (const float*)d_in[6];
    const float* b_u       = (const float*)d_in[7];
    const float* b_r       = (const float*)d_in[8];
    const float* b_c       = (const float*)d_in[9];
    const float* W_u       = (const float*)d_in[10];
    const float* W_r       = (const float*)d_in[11];
    const float* W_c       = (const float*)d_in[12];
    const float* R_p       = (const float*)d_in[13];

    float* outA   = (float*)d_out;                       // [B,N,N]
    float* out_hp = outA + (size_t)BB * NN * NN;         // [B,N,HID]

    unsigned char* ws = (unsigned char*)d_ws;
    unsigned short* kernelT = (unsigned short*)(ws + 0);        //  16 KB
    unsigned short* Wt      = (unsigned short*)(ws + 16384);    // 192 KB
    unsigned short* Rt      = (unsigned short*)(ws + 212992);   //  32 KB
    unsigned short* xpB     = (unsigned short*)(ws + 245760);   //   1 MB
    float*          es      = (float*)        (ws + 1294336);   //  64 KB
    float*          en      = (float*)        (ws + 1359872);   //  64 KB
    unsigned short* z       = (unsigned short*)(ws + 1425408);  //   2 MB
    unsigned short* hpbf    = (unsigned short*)(ws + 3522560);  //   1 MB
    unsigned short* g       = (unsigned short*)(ws + 4571136);  //   1 MB

    k_prep  <<<480, 256, 0, stream>>>(kern, W_u, W_r, W_c, R_p, kernelT, Wt, Rt);
    k_xp    <<<(BB * NN) / 16, 128, 0, stream>>>(x, kernelT, xpB);
    k_scores<<<(BB * HH * NN + 255) / 256, 256, 0, stream>>>(xpB, attn_self, attn_nei, es, en);
    k_hz    <<<(BB * NN * HID + 255) / 256, 256, 0, stream>>>(h, z);
    k_attn  <<<dim3(NN / 16, BB), 128, 0, stream>>>(a, es, en, bias_gat, xpB, z);
    k_gru   <<<dim3(NN / 16, BB), 128, 0, stream>>>(z, Wt, b_u, b_r, b_c, h, out_hp, hpbf);
    k_gp    <<<dim3(NN / 16, BB), 128, 0, stream>>>(hpbf, Rt, g);
    k_dec   <<<dim3(NN / 16, NN / 64, BB), 128, 0, stream>>>(g, hpbf, outA);
}
